// GAT1_5875515261612
// MI455X (gfx1250) — matrix operand order repaired and checked by reading, not compiled, not run on hardware
//
#include <hip/hip_runtime.h>
#include <hip/hip_bf16.h>

typedef __attribute__((ext_vector_type(16))) _Float16 v16h;
typedef __attribute__((ext_vector_type(8)))  _Float16 v8h;
typedef __attribute__((ext_vector_type(8)))  float    v8f;

#define NEG_SLOPE 0.2f
#define ENC_NEG_INF 0x007FFFFFu   // enc(-inf)

// Monotone order-preserving float<->uint32 encoding (for atomic max on floats,
// including negatives): f>=0 -> u|0x80000000 ; f<0 -> ~u.
__device__ __forceinline__ unsigned enc_f32(float f) {
  unsigned u = __float_as_uint(f);
  return (u & 0x80000000u) ? ~u : (u | 0x80000000u);
}
__device__ __forceinline__ float dec_f32(unsigned u) {
  unsigned v = (u & 0x80000000u) ? (u & 0x7FFFFFFFu) : ~u;
  return __uint_as_float(v);
}

__global__ void fill_u32_kernel(unsigned* p, unsigned v, long n) {
  long i = (long)blockIdx.x * blockDim.x + threadIdx.x;
  if (i < n) p[i] = v;
}

__global__ void cvt_f32_f16_kernel(const float* __restrict__ in, _Float16* __restrict__ out, long n) {
  long i = (long)blockIdx.x * blockDim.x + threadIdx.x;
  if (i < n) out[i] = (_Float16)in[i];
}

// out[n*K + k] = (f16) in[k*N + n]  — row-major [K,N] f32 -> col-major (i.e. [N,K]) f16.
__global__ void transpose_f32_f16_kernel(const float* __restrict__ in, _Float16* __restrict__ out,
                                         int K, int N) {
  int i = blockIdx.x * blockDim.x + threadIdx.x;
  if (i >= K * N) return;
  int n = i / K, k = i % K;
  out[i] = (_Float16)in[(long)k * N + n];
}

// C[M,N] = A[M,64] * B[64,N]; A row-major f16, Bt = B^T ([N,64] row-major) f16, C f32.
// M multiple of 16, N-tiles multiple of 4 (4 waves per 128-thread block).
// One wave computes one 16x16 tile via two v_wmma_f32_16x16x32_f16.
__global__ void wmma_gemm_kernel(const _Float16* __restrict__ A, const _Float16* __restrict__ Bt,
                                 float* __restrict__ C, int N) {
  constexpr int K = 64;
  int lane = threadIdx.x & 31;
  int wv   = threadIdx.x >> 5;
  int tm   = blockIdx.x;
  int tn   = blockIdx.y * 4 + wv;
  int hf   = lane >> 4;     // half-wave select
  int l15  = lane & 15;

  // A 16x32 f16 fragment layout: lane (hf,l15) holds row l15,
  //   elems 0..7  -> K = k0 + hf*16 + {0..7}
  //   elems 8..15 -> K = k0 + hf*16 + 8 + {0..7}
  const _Float16* Arow = A + (long)(tm * 16 + l15) * K + hf * 16;
  // B 32x16 f16 fragment layout: lane (hf,l15) holds col l15, K = k0 + hf*16 + {0..15}
  // With Bt col-major these 16 halfs are contiguous.
  const _Float16* Bcol = Bt + (long)(tn * 16 + l15) * K + hf * 16;

  v8f c = {};
#pragma unroll
  for (int k0 = 0; k0 < K; k0 += 32) {
    v8h a0 = *(const v8h*)(Arow + k0);
    v8h a1 = *(const v8h*)(Arow + k0 + 8);
    v16h a = __builtin_shufflevector(a0, a1, 0, 1, 2, 3, 4, 5, 6, 7,
                                     8, 9, 10, 11, 12, 13, 14, 15);
    v16h b = *(const v16h*)(Bcol + k0);
    c = __builtin_amdgcn_wmma_f32_16x16x32_f16(false, a, false, b, (short)0, c, false, false);
  }
  // D layout: vgpr v -> row = tm*16 + hf*8 + v, col = tn*16 + l15
  float* Cp = C + (long)(tm * 16 + hf * 8) * N + tn * 16 + l15;
#pragma unroll
  for (int v = 0; v < 8; ++v) Cp[(long)v * N] = c[v];
}

// el[n,h] = sum_d f[n,h,d]*al[h,d]; er likewise. One thread per (n,h). D multiple of 4.
__global__ void attn_scores_kernel(const float* __restrict__ f, const float* __restrict__ al,
                                   const float* __restrict__ ar, float* __restrict__ el,
                                   float* __restrict__ er, int NH, int H, int D) {
  int i = blockIdx.x * blockDim.x + threadIdx.x;
  if (i >= NH) return;
  int h = i % H;
  const float4* fp  = (const float4*)(f + (long)i * D);
  const float4* alp = (const float4*)(al + (long)h * D);
  const float4* arp = (const float4*)(ar + (long)h * D);
  float sl = 0.f, sr = 0.f;
  int nq = D >> 2;
  for (int q = 0; q < nq; ++q) {
    float4 v = fp[q], a = alp[q], r = arp[q];
    sl += v.x * a.x + v.y * a.y + v.z * a.z + v.w * a.w;
    sr += v.x * r.x + v.y * r.y + v.z * r.z + v.w * r.w;
  }
  el[i] = sl; er[i] = sr;
}

__device__ __forceinline__ float lrelu(float z) { return z > 0.f ? z : NEG_SLOPE * z; }

// Per-(edge,head): segment max of leaky-relu logits over dst (encoded uint atomic max).
__global__ void edge_max_kernel(const int* __restrict__ src, const int* __restrict__ dst,
                                const float* __restrict__ el, const float* __restrict__ er,
                                unsigned* __restrict__ menc, long EH, int H) {
  long i = (long)blockIdx.x * blockDim.x + threadIdx.x;
  if (i >= EH) return;
  int e = (int)(i / H), h = (int)(i % H);
  int s = src[e], d = dst[e];
  float z = lrelu(el[(long)s * H + h] + er[(long)d * H + h]);
  atomicMax(&menc[(long)d * H + h], enc_f32(z));
}

// ex = exp(z - m[dst]); store ex, atomic-add into s[dst].
__global__ void edge_expsum_kernel(const int* __restrict__ src, const int* __restrict__ dst,
                                   const float* __restrict__ el, const float* __restrict__ er,
                                   const unsigned* __restrict__ menc, float* __restrict__ exbuf,
                                   float* __restrict__ ssum, long EH, int H) {
  long i = (long)blockIdx.x * blockDim.x + threadIdx.x;
  if (i >= EH) return;
  int e = (int)(i / H), h = (int)(i % H);
  int s = src[e], d = dst[e];
  float z = lrelu(el[(long)s * H + h] + er[(long)d * H + h]);
  float m = dec_f32(menc[(long)d * H + h]);
  if (!(m > -3.0e38f)) m = 0.f;   // isfinite safety (matches reference where())
  float ex = __expf(z - m);
  exbuf[i] = ex;
  atomicAdd(&ssum[(long)d * H + h], ex);
}

// out[dst,h,:] += f[src,h,:] * (ex/s[dst,h]) — atomic float adds, float4 gathers.
__global__ void edge_aggregate_kernel(const int* __restrict__ src, const int* __restrict__ dst,
                                      const float* __restrict__ f, const float* __restrict__ exbuf,
                                      const float* __restrict__ ssum, float* __restrict__ out,
                                      long EH, int H, int D) {
  long i = (long)blockIdx.x * blockDim.x + threadIdx.x;
  if (i >= EH) return;
  int e = (int)(i / H), h = (int)(i % H);
  int s = src[e], d = dst[e];
  float a = exbuf[i] / ssum[(long)d * H + h];
  const float4* fp = (const float4*)(f + ((long)s * H + h) * D);
  float* op = out + ((long)d * H + h) * D;
  int nq = D >> 2;
  for (int j = 0; j < nq; ++j) {
    float4 v = fp[j];
    atomicAdd(&op[4 * j + 0], v.x * a);
    atomicAdd(&op[4 * j + 1], v.y * a);
    atomicAdd(&op[4 * j + 2], v.z * a);
    atomicAdd(&op[4 * j + 3], v.w * a);
  }
}

// Layer1 epilogue: h[n,d] = sum_h relu(agg[n,h,d] + b[h,d]); emit f16 for next GEMM.
__global__ void finalize1_kernel(const float* __restrict__ agg, const float* __restrict__ b,
                                 _Float16* __restrict__ hh, int N, int H, int D) {
  int i = blockIdx.x * blockDim.x + threadIdx.x;
  if (i >= N * D) return;
  int n = i / D, d = i % D;
  float acc = 0.f;
  for (int h = 0; h < H; ++h) {
    float v = agg[((long)n * H + h) * D + d] + b[(long)h * D + d];
    acc += v > 0.f ? v : 0.f;
  }
  hh[i] = (_Float16)acc;
}

// Layer2 epilogue fused with graph max-readout (encoded atomic max).
__global__ void finalize2_readout_kernel(const float* __restrict__ agg, const float* __restrict__ b,
                                         const int* __restrict__ gid, unsigned* __restrict__ genc,
                                         int N, int D) {
  int i = blockIdx.x * blockDim.x + threadIdx.x;
  if (i >= N * D) return;
  int n = i / D, d = i % D;
  float v = agg[i] + b[d];
  v = v > 0.f ? v : 0.f;
  atomicMax(&genc[(long)gid[n] * D + d], enc_f32(v));
}

// Per-graph MLP: relu(g@lw1+lb1) then relu(·@lw2+lb2). 1 block / graph, 128 threads, LDS.
__global__ void mlp_kernel(const unsigned* __restrict__ genc, const float* __restrict__ lw1,
                           const float* __restrict__ lb1, const float* __restrict__ lw2,
                           const float* __restrict__ lb2, float* __restrict__ out) {
  __shared__ float sg[128];
  __shared__ float red[128];
  int b = blockIdx.x, t = threadIdx.x;
  float g = dec_f32(genc[(long)b * 128 + t]);
  if (!(g > -3.0e38f)) g = 0.f;   // empty-graph safety
  sg[t] = g;
  __syncthreads();
  float acc = lb1[t];
  for (int d = 0; d < 128; ++d) acc += sg[d] * lw1[(long)d * 128 + t];
  acc = acc > 0.f ? acc : 0.f;
  red[t] = acc * lw2[t];
  __syncthreads();
  for (int s = 64; s > 0; s >>= 1) { if (t < s) red[t] += red[t + s]; __syncthreads(); }
  if (t == 0) { float o = red[0] + lb2[0]; out[b] = o > 0.f ? o : 0.f; }
}

static inline int cdiv(long a, long b) { return (int)((a + b - 1) / b); }

extern "C" void kernel_launch(void* const* d_in, const int* in_sizes, int n_in,
                              void* d_out, int out_size, void* d_ws, size_t ws_size,
                              hipStream_t stream) {
  (void)in_sizes; (void)n_in; (void)out_size; (void)ws_size;
  const float* x   = (const float*)d_in[0];
  const int*   src = (const int*)d_in[1];
  const int*   dst = (const int*)d_in[2];
  const int*   gid = (const int*)d_in[3];
  const float* W1  = (const float*)d_in[4];
  const float* al1 = (const float*)d_in[5];
  const float* ar1 = (const float*)d_in[6];
  const float* b1  = (const float*)d_in[7];
  const float* W2  = (const float*)d_in[8];
  const float* al2 = (const float*)d_in[9];
  const float* ar2 = (const float*)d_in[10];
  const float* b2  = (const float*)d_in[11];
  const float* lw1 = (const float*)d_in[12];
  const float* lb1 = (const float*)d_in[13];
  const float* lw2 = (const float*)d_in[14];
  const float* lb2 = (const float*)d_in[15];
  float* out = (float*)d_out;

  const int N = 20000, E = 320000, B = 100;
  const int Din = 64, H1 = 10, D1 = 64, HD1 = 640, D2 = 128;

  char* ws = (char*)d_ws;
  size_t off = 0;
  auto alloc = [&](size_t bytes) -> char* {
    char* p = ws + off;
    off += (bytes + 255) & ~(size_t)255;
    return p;
  };
  _Float16* xh    = (_Float16*)alloc((size_t)N * Din * 2);
  _Float16* W1t   = (_Float16*)alloc((size_t)Din * HD1 * 2);  // [640,64] = W1^T, f16
  _Float16* W2t   = (_Float16*)alloc((size_t)Din * D2 * 2);   // [128,64] = W2^T, f16
  float*    f1    = (float*)alloc((size_t)N * HD1 * 4);
  float*    el1   = (float*)alloc((size_t)N * H1 * 4);
  float*    er1   = (float*)alloc((size_t)N * H1 * 4);
  unsigned* menc1 = (unsigned*)alloc((size_t)N * H1 * 4);
  float*    ssum1 = (float*)alloc((size_t)N * H1 * 4);
  float*    ex1   = (float*)alloc((size_t)E * H1 * 4);
  float*    agg1  = (float*)alloc((size_t)N * HD1 * 4);
  _Float16* hh    = (_Float16*)alloc((size_t)N * D1 * 2);
  float*    f2    = (float*)alloc((size_t)N * D2 * 4);
  float*    el2   = (float*)alloc((size_t)N * 4);
  float*    er2   = (float*)alloc((size_t)N * 4);
  unsigned* menc2 = (unsigned*)alloc((size_t)N * 4);
  float*    ssum2 = (float*)alloc((size_t)N * 4);
  float*    ex2   = (float*)alloc((size_t)E * 4);
  float*    agg2  = (float*)alloc((size_t)N * D2 * 4);
  unsigned* genc  = (unsigned*)alloc((size_t)B * D2 * 4);

  const int T = 256;
  // ---- input conversion + weight transpose (f32 -> f16)
  cvt_f32_f16_kernel<<<cdiv((long)N * Din, T), T, 0, stream>>>(x, xh, (long)N * Din);
  transpose_f32_f16_kernel<<<cdiv((long)Din * HD1, T), T, 0, stream>>>(W1, W1t, Din, HD1);
  transpose_f32_f16_kernel<<<cdiv((long)Din * D2, T), T, 0, stream>>>(W2, W2t, Din, D2);

  // ---- Layer 1: GEMM (1250 x 40 tiles), attention, softmax, aggregate
  wmma_gemm_kernel<<<dim3(N / 16, HD1 / 64), 128, 0, stream>>>(xh, W1t, f1, HD1);
  attn_scores_kernel<<<cdiv((long)N * H1, T), T, 0, stream>>>(f1, al1, ar1, el1, er1, N * H1, H1, D1);
  fill_u32_kernel<<<cdiv((long)N * H1, T), T, 0, stream>>>(menc1, ENC_NEG_INF, (long)N * H1);
  fill_u32_kernel<<<cdiv((long)N * H1, T), T, 0, stream>>>((unsigned*)ssum1, 0u, (long)N * H1);
  fill_u32_kernel<<<cdiv((long)N * HD1, T), T, 0, stream>>>((unsigned*)agg1, 0u, (long)N * HD1);
  edge_max_kernel<<<cdiv((long)E * H1, T), T, 0, stream>>>(src, dst, el1, er1, menc1, (long)E * H1, H1);
  edge_expsum_kernel<<<cdiv((long)E * H1, T), T, 0, stream>>>(src, dst, el1, er1, menc1, ex1, ssum1, (long)E * H1, H1);
  edge_aggregate_kernel<<<cdiv((long)E * H1, T), T, 0, stream>>>(src, dst, f1, ex1, ssum1, agg1, (long)E * H1, H1, D1);
  finalize1_kernel<<<cdiv((long)N * D1, T), T, 0, stream>>>(agg1, b1, hh, N, H1, D1);

  // ---- Layer 2: GEMM (1250 x 8 tiles), attention, softmax, aggregate, readout
  wmma_gemm_kernel<<<dim3(N / 16, D2 / 64), 128, 0, stream>>>(hh, W2t, f2, D2);
  attn_scores_kernel<<<cdiv((long)N, T), T, 0, stream>>>(f2, al2, ar2, el2, er2, N, 1, D2);
  fill_u32_kernel<<<cdiv((long)N, T), T, 0, stream>>>(menc2, ENC_NEG_INF, (long)N);
  fill_u32_kernel<<<cdiv((long)N, T), T, 0, stream>>>((unsigned*)ssum2, 0u, (long)N);
  fill_u32_kernel<<<cdiv((long)N * D2, T), T, 0, stream>>>((unsigned*)agg2, 0u, (long)N * D2);
  fill_u32_kernel<<<cdiv((long)B * D2, T), T, 0, stream>>>(genc, ENC_NEG_INF, (long)B * D2);
  edge_max_kernel<<<cdiv((long)E, T), T, 0, stream>>>(src, dst, el2, er2, menc2, (long)E, 1);
  edge_expsum_kernel<<<cdiv((long)E, T), T, 0, stream>>>(src, dst, el2, er2, menc2, ex2, ssum2, (long)E, 1);
  edge_aggregate_kernel<<<cdiv((long)E, T), T, 0, stream>>>(src, dst, f2, ex2, ssum2, agg2, (long)E, 1, D2);
  finalize2_readout_kernel<<<cdiv((long)N * D2, T), T, 0, stream>>>(agg2, b2, gid, genc, N, D2);

  // ---- Per-graph MLP head
  mlp_kernel<<<B, 128, 0, stream>>>(genc, lw1, lb1, lw2, lb2, out);
}
